// WindowAttention_60928406061298
// MI455X (gfx1250) — matrix operand order repaired and checked by reading, not compiled, not run on hardware
//
#include <hip/hip_runtime.h>
#include <hip/hip_bf16.h>
#include <math.h>

typedef __attribute__((ext_vector_type(16))) __bf16 v16bf;
typedef __attribute__((ext_vector_type(8)))  __bf16 v8bf;
typedef __attribute__((ext_vector_type(8)))  float  v8f;

#define WMMA_BF16(a,b,c) \
  __builtin_amdgcn_wmma_f32_16x16x32_bf16(false,(a),false,(b),(short)0,(c),false,false)

#if defined(__AMDGCN__) && __has_builtin(__builtin_amdgcn_global_load_async_to_lds_b128)
#define ASYNC_LDS 1
#else
#define ASYNC_LDS 0
#endif

__device__ __forceinline__ __bf16 f2bf(float f) {
  unsigned u = __float_as_uint(f);
  unsigned r = (u + 0x7FFFu + ((u >> 16) & 1u)) >> 16;
  unsigned short s = (unsigned short)r;
  __bf16 o;
  __builtin_memcpy(&o, &s, sizeof(o));
  return o;
}

// 16-byte global -> LDS copy; async (ASYNCcnt-tracked) when available.
typedef __attribute__((vector_size(16))) int i32x4;
typedef __attribute__((address_space(1))) i32x4 gbl_i32x4;
typedef __attribute__((address_space(3))) i32x4 lds_i32x4;

__device__ __forceinline__ void cp16(__bf16* dst_lds, const __bf16* src_glob) {
#if ASYNC_LDS
  __builtin_amdgcn_global_load_async_to_lds_b128(
      (gbl_i32x4*)src_glob, (lds_i32x4*)dst_lds, 0, 0);
#else
  *(uint4*)dst_lds = *(const uint4*)src_glob;
#endif
}

__device__ __forceinline__ void wait_async0() {
#if ASYNC_LDS
#if __has_builtin(__builtin_amdgcn_s_wait_asynccnt)
  __builtin_amdgcn_s_wait_asynccnt(0);
#else
  asm volatile("s_wait_asynccnt 0x0" ::: "memory");
#endif
#endif
}

// A-operand fragment (16x32 bf16, MxK). Lane holds row (lane&15);
// K chunks {base..base+7} and {base+16..base+23}, base = (lane&16)?8:0.
__device__ __forceinline__ v16bf frag_a(const __bf16* tile, int stride, int lane) {
  const __bf16* p = tile + (lane & 15) * stride + ((lane & 16) ? 8 : 0);
  v8bf lo = *(const v8bf*)(p);
  v8bf hi = *(const v8bf*)(p + 16);
  v16bf r;
#pragma unroll
  for (int i = 0; i < 8; ++i) { r[i] = lo[i]; r[i + 8] = hi[i]; }
  return r;
}

// B-operand fragment (32x16 bf16, KxN) from an [N][K] tile. Lane holds
// column (lane&15); K chunks {base..base+7} and {base+16..base+23}, base = (lane&16)?8:0.
__device__ __forceinline__ v16bf frag_b(const __bf16* tile, int stride, int lane) {
  const __bf16* p = tile + (lane & 15) * stride + ((lane & 16) ? 8 : 0);
  v8bf lo = *(const v8bf*)(p);
  v8bf hi = *(const v8bf*)(p + 16);
  v16bf r;
#pragma unroll
  for (int i = 0; i < 8; ++i) { r[i] = lo[i]; r[i + 8] = hi[i]; }
  return r;
}

// ---------------------------------------------------------------------------
// Prep kernels: one-pass conversions so GEMM staging is pure 16B copies.
// ---------------------------------------------------------------------------
__global__ __launch_bounds__(256) void convert_x_kernel(
    const float4* __restrict__ x, __bf16* __restrict__ xb, int n4) {
  int stride = gridDim.x * blockDim.x;
  for (int i = blockIdx.x * blockDim.x + threadIdx.x; i < n4; i += stride) {
    float4 f = x[i];
    union { __bf16 h[4]; uint2 u; } o;
    o.h[0] = f2bf(f.x); o.h[1] = f2bf(f.y); o.h[2] = f2bf(f.z); o.h[3] = f2bf(f.w);
    *(uint2*)(xb + (size_t)i * 4) = o.u;
  }
}

// wT[n][k] = bf16(w[k][n]);  K = 256 fixed, N passed in.
__global__ __launch_bounds__(256) void transpose_w_kernel(
    const float* __restrict__ w, __bf16* __restrict__ wT, int N) {
  int t = blockIdx.x * blockDim.x + threadIdx.x;   // t = n*256 + k
  int n = t >> 8, k = t & 255;
  wT[(size_t)n * 256 + k] = f2bf(w[(size_t)k * N + n]);
}

// biasw[h][l][m] = rel_bias[relative_index[l*49+m]][h]
__global__ __launch_bounds__(256) void bias_gather_kernel(
    const int* __restrict__ relative_index, const float* __restrict__ rel_bias,
    float* __restrict__ biasw) {
  int t = blockIdx.x * blockDim.x + threadIdx.x;
  if (t < 8 * 49 * 49) {
    int h = t / 2401, r = t - h * 2401;
    biasw[t] = rel_bias[relative_index[r] * 8 + h];
  }
}

// ---------------------------------------------------------------------------
// Kernel 1: qkv = xb @ wqkvT^T + qkv_b (M=200704, N=768, K=256), bf16 out,
// scattered into q:[B,H,L,32], k:[B,H,L,32], v:[B,H,32,64] (L padded ->64).
// Double-buffered async LDS staging.
// ---------------------------------------------------------------------------
__global__ __launch_bounds__(256) void qkv_gemm_kernel(
    const __bf16* __restrict__ xb, const __bf16* __restrict__ wT,
    const float* __restrict__ qkv_b,
    __bf16* __restrict__ qws, __bf16* __restrict__ kws, __bf16* __restrict__ vws) {
  __shared__ __align__(16) __bf16 As[2][128][32];   // [m][k]
  __shared__ __align__(16) __bf16 Bs[2][128][40];   // [n][k], pad keeps 16B align

  const int tid  = threadIdx.x;
  const int lane = tid & 31;
  const int w    = tid >> 5;
  const int wm   = w & 3;
  const int wn   = w >> 2;
  const int m0   = blockIdx.x * 128;
  const int n0   = blockIdx.y * 128;

  v8f acc[2][4];
  v8f zero8 = {0.f,0.f,0.f,0.f,0.f,0.f,0.f,0.f};
#pragma unroll
  for (int i = 0; i < 2; ++i)
#pragma unroll
    for (int j = 0; j < 4; ++j) acc[i][j] = zero8;

  const int sr = tid >> 1, scs = (tid & 1) * 16;   // staging coords
  auto stage = [&](int bsel, int kb) {
    const int k0 = kb * 32;
    cp16(&As[bsel][sr][scs], xb + (size_t)(m0 + sr) * 256 + k0 + scs);
    cp16(&Bs[bsel][sr][scs], wT + (size_t)(n0 + sr) * 256 + k0 + scs);
  };

  stage(0, 0);
  int buf = 0;
  for (int kb = 0; kb < 8; ++kb) {
    wait_async0();
    __syncthreads();
    if (kb < 7) stage(buf ^ 1, kb + 1);
    v16bf a0 = frag_a(&As[buf][wm*32     ][0], 32, lane);
    v16bf a1 = frag_a(&As[buf][wm*32 + 16][0], 32, lane);
#pragma unroll
    for (int jn = 0; jn < 4; ++jn) {
      v16bf bb = frag_b(&Bs[buf][wn*64 + jn*16][0], 40, lane);
      acc[0][jn] = WMMA_BF16(a0, bb, acc[0][jn]);
      acc[1][jn] = WMMA_BF16(a1, bb, acc[1][jn]);
    }
    buf ^= 1;
  }

  const int lrow = (lane & 16) ? 8 : 0;
#pragma unroll
  for (int jn = 0; jn < 4; ++jn) {
    const int n_blk = n0 + wn*64 + jn*16;     // uniform per fragment
    const int which = n_blk >> 8;             // 0=q 1=k 2=v
    const int h     = (n_blk >> 5) & 7;
    const int d     = (n_blk & 31) + (lane & 15);
    const float bias = qkv_b[n_blk + (lane & 15)];
#pragma unroll
    for (int im = 0; im < 2; ++im) {
      int m = m0 + wm*32 + im*16 + lrow;
      unsigned bw = (unsigned)m / 49u;        // 2 divides/lane instead of 16
      unsigned l  = (unsigned)m - bw * 49u;
#pragma unroll
      for (int j = 0; j < 8; ++j) {
        __bf16 val = f2bf(acc[im][jn][j] + bias);
        if (which == 0)      qws[(((size_t)bw*8 + h)*49 + l)*32 + d] = val;
        else if (which == 1) kws[(((size_t)bw*8 + h)*49 + l)*32 + d] = val;
        else                 vws[(((size_t)bw*8 + h)*32 + d)*64 + l] = val;
        ++l; if (l == 49u) { l = 0; ++bw; }
      }
    }
  }
}

// ---------------------------------------------------------------------------
// Kernel 2: per (b,h): S = scale*q@k^T + bias; P = softmax(S); O = P@v.
// ---------------------------------------------------------------------------
__global__ __launch_bounds__(128) void attn_kernel(
    const __bf16* __restrict__ qws, const __bf16* __restrict__ kws,
    const __bf16* __restrict__ vws, const float* __restrict__ biasw,
    __bf16* __restrict__ aows) {
  __shared__ __align__(16) __bf16 qs[64][32];    // q  [l][d]
  __shared__ __align__(16) __bf16 ks[64][32];    // k  [m][d]
  __shared__ __align__(16) __bf16 vts[32][64];   // v^T[d][m]
  __shared__            float  sf[64][65];
  __shared__ __align__(16) __bf16 pt[64][64];

  const int tid  = threadIdx.x;
  const int lane = tid & 31;
  const int w    = tid >> 5;
  const int bh   = blockIdx.x;
  const int h    = bh & 7;

  {   // q/k: 2 chunks/thread; rows >= 49 read neighbor data (rows unused)
    int l = tid >> 1, ds = (tid & 1) * 16;
    size_t base = ((size_t)bh * 49 + l) * 32 + ds;
    cp16(&qs[l][ds], qws + base);
    cp16(&ks[l][ds], kws + base);
  }
  {   // v^T: rows are 128B (L padded to 64) -> pure 16B chunks
    int d = tid >> 2, ms = (tid & 3) * 16;
    cp16(&vts[d][ms], vws + ((size_t)bh * 32 + d) * 64 + ms);
  }
  wait_async0();
  __syncthreads();

  const float scale = 0.17677669529663687f;   // 1/sqrt(32)
  const int lrow = (lane & 16) ? 8 : 0;
  const float* brow = biasw + h * 2401;
  v16bf aq = frag_a(&qs[w*16][0], 32, lane);
  v8f zero8 = {0.f,0.f,0.f,0.f,0.f,0.f,0.f,0.f};
#pragma unroll
  for (int jn = 0; jn < 4; ++jn) {
    v16bf bk = frag_b(&ks[jn*16][0], 32, lane);
    v8f s = WMMA_BF16(aq, bk, zero8);
    int m  = jn*16 + (lane & 15);
    int mi = m < 49 ? m : 48;
#pragma unroll
    for (int j = 0; j < 8; ++j) {
      int l  = w*16 + j + lrow;
      int li = l < 49 ? l : 48;
      sf[l][m] = s[j] * scale + brow[li*49 + mi];
    }
  }
  __syncthreads();

  if (tid < 64) {
    int r = tid;
    if (r < 49) {
      float mx = -1e30f;
      for (int m = 0; m < 49; ++m) mx = fmaxf(mx, sf[r][m]);
      float sum = 0.0f;
      for (int m = 0; m < 49; ++m) {
        float e = __expf(sf[r][m] - mx);
        sf[r][m] = e;
        sum += e;
      }
      float inv = 1.0f / sum;
      for (int m = 0; m < 49; ++m) pt[r][m] = f2bf(sf[r][m] * inv);
      for (int m = 49; m < 64; ++m) pt[r][m] = f2bf(0.0f);
    } else {
      for (int m = 0; m < 64; ++m) pt[r][m] = f2bf(0.0f);
    }
  }
  __syncthreads();

  v8f o0 = zero8, o1 = zero8;
#pragma unroll
  for (int kc = 0; kc < 2; ++kc) {
    v16bf ap  = frag_a(&pt[w*16][kc*32], 64, lane);
    v16bf bv0 = frag_b(&vts[0 ][kc*32], 64, lane);
    v16bf bv1 = frag_b(&vts[16][kc*32], 64, lane);
    o0 = WMMA_BF16(ap, bv0, o0);
    o1 = WMMA_BF16(ap, bv1, o1);
  }
  int b = bh >> 3;
#pragma unroll
  for (int j = 0; j < 8; ++j) {
    int l = w*16 + j + lrow;
    if (l < 49) {
      size_t row = ((size_t)b * 49 + l) * 256 + h * 32;
      aows[row + (lane & 15)]      = f2bf(o0[j]);
      aows[row + 16 + (lane & 15)] = f2bf(o1[j]);
    }
  }
}

// ---------------------------------------------------------------------------
// Kernel 3: out = ao @ wprojT^T + proj_b (M=200704, N=256, K=256), fp32 out.
// ---------------------------------------------------------------------------
__global__ __launch_bounds__(256) void proj_gemm_kernel(
    const __bf16* __restrict__ ain, const __bf16* __restrict__ wT,
    const float* __restrict__ proj_b, float* __restrict__ out) {
  __shared__ __align__(16) __bf16 As[2][128][32];
  __shared__ __align__(16) __bf16 Bs[2][128][40];

  const int tid  = threadIdx.x;
  const int lane = tid & 31;
  const int w    = tid >> 5;
  const int wm   = w & 3;
  const int wn   = w >> 2;
  const int m0   = blockIdx.x * 128;
  const int n0   = blockIdx.y * 128;

  v8f acc[2][4];
  v8f zero8 = {0.f,0.f,0.f,0.f,0.f,0.f,0.f,0.f};
#pragma unroll
  for (int i = 0; i < 2; ++i)
#pragma unroll
    for (int j = 0; j < 4; ++j) acc[i][j] = zero8;

  const int sr = tid >> 1, scs = (tid & 1) * 16;
  auto stage = [&](int bsel, int kb) {
    const int k0 = kb * 32;
    cp16(&As[bsel][sr][scs], ain + (size_t)(m0 + sr) * 256 + k0 + scs);
    cp16(&Bs[bsel][sr][scs], wT + (size_t)(n0 + sr) * 256 + k0 + scs);
  };

  stage(0, 0);
  int buf = 0;
  for (int kb = 0; kb < 8; ++kb) {
    wait_async0();
    __syncthreads();
    if (kb < 7) stage(buf ^ 1, kb + 1);
    v16bf a0 = frag_a(&As[buf][wm*32     ][0], 32, lane);
    v16bf a1 = frag_a(&As[buf][wm*32 + 16][0], 32, lane);
#pragma unroll
    for (int jn = 0; jn < 4; ++jn) {
      v16bf bb = frag_b(&Bs[buf][wn*64 + jn*16][0], 40, lane);
      acc[0][jn] = WMMA_BF16(a0, bb, acc[0][jn]);
      acc[1][jn] = WMMA_BF16(a1, bb, acc[1][jn]);
    }
    buf ^= 1;
  }

  const int lrow = (lane & 16) ? 8 : 0;
#pragma unroll
  for (int jn = 0; jn < 4; ++jn) {
    int n    = n0 + wn*64 + jn*16 + (lane & 15);
    float pb = proj_b[n];
#pragma unroll
    for (int im = 0; im < 2; ++im) {
#pragma unroll
      for (int j = 0; j < 8; ++j) {
        int m = m0 + wm*32 + im*16 + j + lrow;
        out[(size_t)m * 256 + n] = acc[im][jn][j] + pb;
      }
    }
  }
}

// ---------------------------------------------------------------------------
extern "C" void kernel_launch(void* const* d_in, const int* in_sizes, int n_in,
                              void* d_out, int out_size, void* d_ws, size_t ws_size,
                              hipStream_t stream) {
  const float* x              = (const float*)d_in[0];
  const int*   relative_index = (const int*)  d_in[1];
  const float* qkv_w          = (const float*)d_in[2];
  const float* qkv_b          = (const float*)d_in[3];
  const float* rel_bias       = (const float*)d_in[4];
  const float* proj_w         = (const float*)d_in[5];
  const float* proj_b         = (const float*)d_in[6];
  float* out = (float*)d_out;

  const size_t MK   = (size_t)200704 * 256;        // x / ao elements
  const size_t QKE  = (size_t)4096 * 8 * 49 * 32;  // q / k elements
  const size_t VE   = (size_t)4096 * 8 * 32 * 64;  // v (L padded to 64)
  char* ws = (char*)d_ws;
  __bf16* xb     = (__bf16*)(ws);
  __bf16* qws    = (__bf16*)(ws + MK * 2);
  __bf16* kws    = (__bf16*)(ws + MK * 2 + QKE * 2);
  __bf16* vws    = (__bf16*)(ws + MK * 2 + QKE * 4);
  __bf16* aows   = (__bf16*)(ws + MK * 2 + QKE * 4 + VE * 2);
  __bf16* wqkvT  = (__bf16*)(ws + MK * 4 + QKE * 4 + VE * 2);
  __bf16* wprojT = (__bf16*)(ws + MK * 4 + QKE * 4 + VE * 2 + 768 * 256 * 2);
  float*  biasw  = (float*) (ws + MK * 4 + QKE * 4 + VE * 2 + (768 + 256) * 256 * 2);

  convert_x_kernel  <<<4096, 256, 0, stream>>>((const float4*)x, xb, (int)(MK / 4));
  transpose_w_kernel<<<768,  256, 0, stream>>>(qkv_w, wqkvT, 768);
  transpose_w_kernel<<<256,  256, 0, stream>>>(proj_w, wprojT, 256);
  bias_gather_kernel<<<76,   256, 0, stream>>>(relative_index, rel_bias, biasw);

  qkv_gemm_kernel <<<dim3(1568, 6), 256, 0, stream>>>(xb, wqkvT, qkv_b, qws, kws, vws);
  attn_kernel     <<<dim3(4096 * 8), 128, 0, stream>>>(qws, kws, vws, biasw, aows);
  proj_gemm_kernel<<<dim3(1568, 2), 256, 0, stream>>>(aows, wprojT, proj_b, out);
}
